// GLIF3Cupy_27797028340079
// MI455X (gfx1250) — matrix operand order repaired and checked by reading, not compiled, not run on hardware
//
#include <hip/hip_runtime.h>

// ---------------------------------------------------------------------------
// GLIF3 recurrent spiking network, MI455X (gfx1250, wave32, WMMA).
//
// One persistent kernel runs all T=512 steps (device-wide barrier via L2
// atomic counter); per-neuron state lives in registers for the whole run.
// Per step:
//   1) stage s_prev (8 KB) into LDS with GLOBAL_LOAD_ASYNC_TO_LDS_B128
//      (ASYNCcnt path; B-fragments then come from LDS on DScnt, decoupled
//      from the weight stream on LOADcnt),
//   2) lin = w @ s_prev with v_wmma_f32_16x16x32_f16 on f16 weights (32 MB,
//      L2-resident). K split 4-ways across waves (16 waves/WG -> 4 waves per
//      SIMD for latency hiding); partials reduced through LDS in fixed order.
//      All-zero 32-wide spike chunks skipped via a wave-uniform branch (EXEC
//      stays all-1s for WMMA). Weights prefetched into WGP$ with WGP-scope
//      global_prefetch_b8 (inline asm; the builtin only reaches SE scope).
// ---------------------------------------------------------------------------

typedef __attribute__((ext_vector_type(16))) _Float16     v16h;
typedef __attribute__((ext_vector_type(8)))  _Float16     v8h;
typedef __attribute__((ext_vector_type(4)))  _Float16     v4h;
typedef __attribute__((ext_vector_type(8)))  float        v8f;
typedef __attribute__((ext_vector_type(4)))  float        v4f;
typedef __attribute__((ext_vector_type(8)))  unsigned int v8u;

namespace {
constexpr int Tn = 512;
constexpr int Bn = 4096;
constexpr int Mn = 2;
constexpr int NWG          = 64;    // persistent workgroups (well under residency)
constexpr int STEP_THREADS = 512;   // 16 waves
constexpr int ROWS_PER_WG  = 64;    // 4 row-tiles * 16 rows
constexpr int KSLICES      = 4;     // K split across waves
constexpr int KSLICE_LEN   = Bn / KSLICES;  // 1024
}

// ------------------------- setup: w (f32) -> wh (f16) ----------------------
__global__ __launch_bounds__(256) void glif_conv_w(const float* __restrict__ w,
                                                   _Float16* __restrict__ wh) {
  size_t i = ((size_t)blockIdx.x * 256u + threadIdx.x) * 4u;
  v4f f = *(const v4f*)(w + i);
  v4h h;
  h[0] = (_Float16)f[0];
  h[1] = (_Float16)f[1];
  h[2] = (_Float16)f[2];
  h[3] = (_Float16)f[3];
  *(v4h*)(wh + i) = h;
}

// -------------- setup: zero spike ping-pong + barrier counter --------------
__global__ __launch_bounds__(256) void glif_zero(_Float16* __restrict__ sbuf,
                                                 int* __restrict__ counter) {
  int i = blockIdx.x * 256 + threadIdx.x;
  if (i < 2 * Bn) sbuf[i] = (_Float16)0.0f;
  if (i == 0) *counter = 0;
}

// --------------------------- persistent kernel -----------------------------
__global__ __launch_bounds__(STEP_THREADS) void glif_persist(
    const _Float16* __restrict__ wh,     // (B,B) f16 weights
    _Float16*       __restrict__ sbuf,   // (2,B) f16 spike ping-pong
    int*            __restrict__ counter,// device-wide barrier counter
    const float* __restrict__ x_seq,     // (T,B)
    const float* __restrict__ bvec,
    const float* __restrict__ v0,
    const float* __restrict__ Iasc0,     // (B,M)
    const float* __restrict__ vth,
    const float* __restrict__ vreset,
    const float* __restrict__ vrest,
    const float* __restrict__ cm,
    const float* __restrict__ tau,
    const float* __restrict__ kk,        // (B,M)
    const float* __restrict__ asc,       // (B,M)
    const float* __restrict__ maskv,
    float* __restrict__ s_seq,           // (T,B)
    float* __restrict__ v_seq,           // (T,B)
    float* __restrict__ vfin,            // (B,)
    float* __restrict__ Ifin) {          // (B,M)
  __shared__ __attribute__((aligned(64))) _Float16 slds[Bn];  // staged spikes
  __shared__ float lin_part[KSLICES][ROWS_PER_WG];

  const int tid  = threadIdx.x;
  const int lane = tid & 31;
  const int wave = tid >> 5;             // 0..15
  const int tile = wave & 3;             // row tile within WG
  const int ks   = wave >> 2;            // K slice 0..3
  const int wgBase   = blockIdx.x * ROWS_PER_WG;
  const int tileBase = wgBase + tile * 16;
  const int hi   = lane >> 4;            // 0: lanes 0-15, 1: lanes 16-31
  const int row  = tileBase + (lane & 15);
  const int kb0  = ks * KSLICE_LEN;

  // A-fragment K offsets (contiguous, matching B):
  //   lanes 0-15 carry K 0..15, lanes 16-31 carry K 16..31
  const int aoff = hi * 16;
  // B-fragment: lanes 0-15 carry K 0..15, lanes 16-31 carry K 16..31;
  // spike chunk replicated across all 16 N columns.
  const int boff = hi * 16;
  const _Float16* wrow = wh + (size_t)row * Bn;

  // LDS byte offset of this thread's 16B staging slot (low 32 bits of the
  // flat shared address are the in-LDS offset per the aperture mapping).
  const unsigned lds_stage = (unsigned)(uintptr_t)(&slds[0]) + (unsigned)(tid * 16);

  // ---- per-neuron state: registers for the entire sequence ----
  const int i = wgBase + tid;            // valid for tid < 64
  float bv = 0.f, vthr = 1.f, vrs = 0.f, vrt = 0.f, taucm = 0.f, ad = 0.f;
  float bd0 = 0.f, bd1 = 0.f, a0c = 0.f, a1c = 0.f, mk = 0.f, invden = 1.f;
  float vst = 0.f, I0 = 0.f, I1 = 0.f;
  if (tid < ROWS_PER_WG) {
    bv   = bvec[i];
    vthr = vth[i];
    vrs  = vreset[i];
    vrt  = vrest[i];
    taucm = tau[i] / cm[i];
    ad   = __expf(-1.0f / tau[i]);       // DT = 1
    bd0  = __expf(-kk[Mn * i + 0]);
    bd1  = __expf(-kk[Mn * i + 1]);
    a0c  = asc[Mn * i + 0];
    a1c  = asc[Mn * i + 1];
    mk   = maskv[i];
    invden = 1.0f / (vthr - vrs);
    vst  = v0[i];
    I0   = Iasc0[Mn * i + 0];
    I1   = Iasc0[Mn * i + 1];
  }

  for (int t = 0; t < Tn; ++t) {
    const _Float16* sprev = sbuf + (size_t)(t & 1) * Bn;
    _Float16*       scur  = sbuf + (size_t)((t + 1) & 1) * Bn;

    // ---- stage s_prev into LDS: async global->LDS copy (ASYNCcnt) ----
    {
      const _Float16* gsrc = sprev + tid * 8;      // 512 threads * 16B = 8KB
      asm volatile("global_load_async_to_lds_b128 %0, %1, off"
                   :: "v"(lds_stage), "v"(gsrc)
                   : "memory");
      asm volatile("s_wait_asynccnt 0x0" ::: "memory");
    }
    __syncthreads();

    // -------- partial lin = w[tile rows, K-slice] @ s_prev[K-slice] --------
    v8f c = {};
    for (int kb = kb0; kb < kb0 + KSLICE_LEN; kb += 32) {
      v16h bf = *(const v16h*)(&slds[kb + boff]);  // LDS, replicated columns
      v8u  bu = __builtin_bit_cast(v8u, bf);
      unsigned nz = (bu[0] | bu[1]) | (bu[2] | bu[3]) |
                    ((bu[4] | bu[5]) | (bu[6] | bu[7]));
      // Wave-uniform sparsity skip: all-zero spike chunk contributes nothing.
      if (__any(nz != 0u)) {
        // WGP-scope prefetch (fills WGP$; the builtin only reaches SE scope).
        const _Float16* pf = wrow + kb + 256;
        asm volatile("global_prefetch_b8 %0, off" ::"v"(pf));
        v8h a0 = *(const v8h*)(wrow + kb + aoff);
        v8h a1 = *(const v8h*)(wrow + kb + aoff + 8);
        v16h a = __builtin_shufflevector(a0, a1, 0, 1, 2, 3, 4, 5, 6, 7,
                                                 8, 9, 10, 11, 12, 13, 14, 15);
        c = __builtin_amdgcn_wmma_f32_16x16x32_f16(
            false, a, false, bf, (short)0, c, false, false);
      }
    }

    // lane 0: c[r] = partial lin[tileBase+r]; lane 16: rows tileBase+8+r
    if ((lane & 15) == 0) {
#pragma unroll
      for (int r = 0; r < 8; ++r)
        lin_part[ks][tile * 16 + hi * 8 + r] = c[r];
    }
    __syncthreads();

    // ---------------- elementwise GLIF update ----------------
    if (tid < ROWS_PER_WG) {
      // Fixed-order reduction of the 4 K-slice partials (deterministic).
      const float lin = (lin_part[0][tid] + lin_part[1][tid]) +
                        (lin_part[2][tid] + lin_part[3][tid]);
      const float x     = x_seq[(size_t)t * Bn + i] + bv + lin;
      const float vinf  = vrt + taucm * (x + I0 + I1);
      const float vp    = vinf + (vst - vinf) * ad;
      const float u     = (vp - vthr) * invden;
      const float s     = (u > 0.0f) ? mk : 0.0f;
      vst = vp - (vp - vrs) * s;                    // HARD_RESET
      I0  = I0 * bd0 + a0c * s;
      I1  = I1 * bd1 + a1c * s;
      s_seq[(size_t)t * Bn + i] = s;
      v_seq[(size_t)t * Bn + i] = vst;
      scur[i] = (_Float16)s;
    }

    // ---------------- device-wide step barrier ----------------
    __threadfence();                                // release our stores to L2
    __syncthreads();                                // also guards slds/lin_part reuse
    if (tid == 0) {
      atomicAdd(counter, 1);
      const int target = NWG * (t + 1);
      while (__hip_atomic_load(counter, __ATOMIC_RELAXED,
                               __HIP_MEMORY_SCOPE_AGENT) < target) {
        __builtin_amdgcn_s_sleep(1);
      }
    }
    __syncthreads();
    __threadfence();                                // acquire others' spikes
  }

  if (tid < ROWS_PER_WG) {
    vfin[i] = vst;
    Ifin[Mn * i + 0] = I0;
    Ifin[Mn * i + 1] = I1;
  }
}

// ---------------------------------------------------------------------------
extern "C" void kernel_launch(void* const* d_in, const int* in_sizes, int n_in,
                              void* d_out, int out_size, void* d_ws, size_t ws_size,
                              hipStream_t stream) {
  (void)in_sizes; (void)n_in; (void)out_size; (void)ws_size;

  const float* x_seq   = (const float*)d_in[0];   // (T,B)
  const float* w       = (const float*)d_in[1];   // (B,B)
  const float* bvec    = (const float*)d_in[2];
  const float* v0      = (const float*)d_in[3];
  const float* Iasc0   = (const float*)d_in[4];   // (B,M)
  const float* v_th    = (const float*)d_in[5];
  const float* v_reset = (const float*)d_in[6];
  const float* v_rest  = (const float*)d_in[7];
  const float* c_m     = (const float*)d_in[8];
  const float* tau     = (const float*)d_in[9];
  const float* kk      = (const float*)d_in[10];  // (B,M)
  const float* asc     = (const float*)d_in[11];  // (B,M)
  const float* maskv   = (const float*)d_in[12];

  // ---- workspace layout (~32.02 MB) ----
  char*  ws  = (char*)d_ws;
  size_t off = 0;
  auto alloc = [&](size_t bytes) -> void* {
    void* p = ws + off;
    off = (off + bytes + 255) & ~(size_t)255;
    return p;
  };
  _Float16* wh      = (_Float16*)alloc((size_t)Bn * Bn * sizeof(_Float16));
  _Float16* sbuf    = (_Float16*)alloc((size_t)2 * Bn * sizeof(_Float16));
  int*      counter = (int*)alloc(sizeof(int));

  // ---- output layout: s_seq | v_seq | v_fin | I_fin ----
  float* s_seq = (float*)d_out;
  float* v_seq = s_seq + (size_t)Tn * Bn;
  float* vfin  = v_seq + (size_t)Tn * Bn;
  float* Ifin  = vfin + Bn;

  // setup (re-run every call -> deterministic; no cross-call state)
  glif_conv_w<<<(unsigned)((size_t)Bn * Bn / 4 / 256), 256, 0, stream>>>(w, wh);
  glif_zero<<<(2 * Bn + 255) / 256, 256, 0, stream>>>(sbuf, counter);

  glif_persist<<<NWG, STEP_THREADS, 0, stream>>>(
      wh, sbuf, counter,
      x_seq, bvec, v0, Iasc0, v_th, v_reset, v_rest, c_m, tau,
      kk, asc, maskv,
      s_seq, v_seq, vfin, Ifin);
}
